// SoftContrastiveLoss_20512763806439
// MI455X (gfx1250) — matrix operand order repaired and checked by reading, not compiled, not run on hardware
//
#include <hip/hip_runtime.h>
#include <math.h>

typedef __attribute__((ext_vector_type(16))) _Float16 v16h;
typedef __attribute__((ext_vector_type(8)))  _Float16 v8h;
typedef __attribute__((ext_vector_type(8)))  float    v8f;

#define ALPHA 10.0f
#define BETA  2.0f
#define BASE  0.7f
#define SIM_ONE 0.999999f     // 1.0 - 1e-6
#define COLCHUNKS 16          // column-dimension split for parallelism
#define LDS_STRIDE 264        // 256 + 8 halves pad: 528B row -> 4-bank shift/col, 16B aligned

__device__ __forceinline__ float softplus_fast(float z) {
    // stable log(1+exp(z)) using hardware transcendentals
    const float az = fabsf(z);
    const float l  = __logf(1.0f + __expf(-az));
    return (z > 0.0f) ? (z + l) : l;
}

__device__ __forceinline__ void async_cp16(unsigned lds_off, const _Float16* g) {
    asm volatile("global_load_async_to_lds_b128 %0, %1, off"
                 :: "v"(lds_off), "v"(g) : "memory");
}
__device__ __forceinline__ void wait_async() {
    asm volatile("s_wait_asynccnt 0" ::: "memory");
}

// Stage one B panel (16 cols x d halves) into LDS via async-to-LDS copies.
__device__ __forceinline__ void stage_panel(_Float16 (*buf)[LDS_STRIDE],
                                            const _Float16* __restrict__ xh,
                                            int colbase, int d, int kch, int tid) {
    const int totch = 16 * kch;                 // 16B chunks per panel
    for (int c = tid; c < totch; c += 256) {
        const int col = c / kch;
        const int kp  = (c - col * kch) * 8;    // 8 halves per 16B chunk
        const unsigned lds = (unsigned)(size_t)&buf[col][kp];
        async_cp16(lds, xh + (size_t)(colbase + col) * d + kp);
    }
}

// ---------------- f32 -> f16 conversion ----------------
__global__ void cvt_kernel(const float* __restrict__ x, _Float16* __restrict__ xh, size_t nelem) {
    size_t i = (size_t)blockIdx.x * blockDim.x + threadIdx.x;
    if (i < nelem) xh[i] = (_Float16)x[i];
}

// ---------------- Pass 1: per-row min_pos / max_neg (partial over a col chunk) ----
// 8 waves/WG, one 16-row tile per wave (A resident in 64 VGPRs, no spills).
// B panels staged in LDS once per WG via async copies, double buffered.
__global__ void __launch_bounds__(256)
pass1_kernel(const _Float16* __restrict__ xh, const int* __restrict__ tgt,
             float* __restrict__ minpos_p, float* __restrict__ maxneg_p,
             int n, int d) {
    __shared__ _Float16 bbuf[2][16][LDS_STRIDE];

    const int tid  = threadIdx.x;
    const int wave = tid >> 5;
    const int lane = tid & 31;
    const int half = lane >> 4;
    const int q    = lane & 15;
    const int rowbase = (blockIdx.x * 8 + wave) * 16;
    const int chunk   = blockIdx.y;
    const int ntiles  = n >> 4;
    const int tpc     = ntiles / COLCHUNKS;
    const int c0 = chunk * tpc, c1 = c0 + tpc;
    const int kch = d >> 3;

    v16h afrag[8];
    {
        const _Float16* xr = xh + (size_t)(rowbase + q) * d;
#pragma unroll
        for (int t = 0; t < 8; ++t) {
            v8h lo = *(const v8h*)(xr + t * 32 + half * 8);
            v8h hi = *(const v8h*)(xr + t * 32 + 16 + half * 8);
            afrag[t] = __builtin_shufflevector(lo, hi, 0,1,2,3,4,5,6,7,8,9,10,11,12,13,14,15);
        }
    }

    int   trow[8];
    float mn[8], mx[8];
#pragma unroll
    for (int r = 0; r < 8; ++r) {
        trow[r] = tgt[rowbase + half * 8 + r];
        mn[r] = INFINITY; mx[r] = -INFINITY;
    }

    stage_panel(bbuf[0], xh, c0 * 16, d, kch, tid);
    wait_async();
    __syncthreads();

    for (int ct = c0; ct < c1; ++ct) {
        const int cur = (ct - c0) & 1;
        if (ct + 1 < c1) stage_panel(bbuf[cur ^ 1], xh, (ct + 1) * 16, d, kch, tid);

        const int colbase = ct * 16;
        const int tc = tgt[colbase + q];

        v8f c = {};
#pragma unroll
        for (int t = 0; t < 8; ++t) {
            const _Float16* bp = &bbuf[cur][q][t * 32 + half * 8];
            v8h b0 = *(const v8h*)bp;
            v8h b1 = *(const v8h*)(bp + 16);
            v16h b = __builtin_shufflevector(b0, b1, 0,1,2,3,4,5,6,7,8,9,10,11,12,13,14,15);
            c = __builtin_amdgcn_wmma_f32_16x16x32_f16(false, afrag[t], false, b,
                                                       (short)0, c, false, false);
        }

#pragma unroll
        for (int r = 0; r < 8; ++r) {
            const float v = c[r];
            const int i = rowbase + half * 8 + r;
            const int j = colbase + q;
            const bool same = (tc == trow[r]);
            const bool posm = same && (j != i) && (v < SIM_ONE);
            mn[r] = fminf(mn[r], posm ? v : INFINITY);
            mx[r] = fmaxf(mx[r], same ? -INFINITY : v);
        }

        wait_async();
        __syncthreads();
    }

#pragma unroll
    for (int r = 0; r < 8; ++r) {
        float a = mn[r], b = mx[r];
#pragma unroll
        for (int m = 8; m >= 1; m >>= 1) {
            a = fminf(a, __shfl_xor(a, m, 16));
            b = fmaxf(b, __shfl_xor(b, m, 16));
        }
        if (q == 0) {
            const int i = rowbase + half * 8 + r;
            minpos_p[(size_t)chunk * n + i] = a;
            maxneg_p[(size_t)chunk * n + i] = b;
        }
    }
}

// ---------------- combine pass-1 partials ----------------
__global__ void combine_minmax(const float* __restrict__ mp_p, const float* __restrict__ mx_p,
                               float* __restrict__ minpos, float* __restrict__ maxneg, int n) {
    const int i = blockIdx.x * blockDim.x + threadIdx.x;
    if (i < n) {
        float a = INFINITY, b = -INFINITY;
        for (int c = 0; c < COLCHUNKS; ++c) {
            a = fminf(a, mp_p[(size_t)c * n + i]);
            b = fmaxf(b, mx_p[(size_t)c * n + i]);
        }
        minpos[i] = a; maxneg[i] = b;
    }
}

// ---------------- Pass 2: hard-mining + softplus partial sums ----------------
__global__ void __launch_bounds__(256)
pass2_kernel(const _Float16* __restrict__ xh, const int* __restrict__ tgt,
             const float* __restrict__ minpos_g, const float* __restrict__ maxneg_g,
             const float* __restrict__ margin_p,
             float* __restrict__ ps_p, float* __restrict__ pc_p,
             float* __restrict__ ns_p, float* __restrict__ nc_p,
             int n, int d) {
    __shared__ _Float16 bbuf[2][16][LDS_STRIDE];

    const int tid  = threadIdx.x;
    const int wave = tid >> 5;
    const int lane = tid & 31;
    const int half = lane >> 4;
    const int q    = lane & 15;
    const int rowbase = (blockIdx.x * 8 + wave) * 16;
    const int chunk   = blockIdx.y;
    const int ntiles  = n >> 4;
    const int tpc     = ntiles / COLCHUNKS;
    const int c0 = chunk * tpc, c1 = c0 + tpc;
    const int kch = d >> 3;
    const float margin = *margin_p;

    v16h afrag[8];
    {
        const _Float16* xr = xh + (size_t)(rowbase + q) * d;
#pragma unroll
        for (int t = 0; t < 8; ++t) {
            v8h lo = *(const v8h*)(xr + t * 32 + half * 8);
            v8h hi = *(const v8h*)(xr + t * 32 + 16 + half * 8);
            afrag[t] = __builtin_shufflevector(lo, hi, 0,1,2,3,4,5,6,7,8,9,10,11,12,13,14,15);
        }
    }

    int   trow[8];
    float mp[8], mxg[8];
    float ps[8], pc[8], ns[8], nc[8];
#pragma unroll
    for (int r = 0; r < 8; ++r) {
        const int i = rowbase + half * 8 + r;
        trow[r] = tgt[i];
        mp[r]   = minpos_g[i];
        mxg[r]  = maxneg_g[i];
        ps[r] = 0.f; pc[r] = 0.f; ns[r] = 0.f; nc[r] = 0.f;
    }

    stage_panel(bbuf[0], xh, c0 * 16, d, kch, tid);
    wait_async();
    __syncthreads();

    for (int ct = c0; ct < c1; ++ct) {
        const int cur = (ct - c0) & 1;
        if (ct + 1 < c1) stage_panel(bbuf[cur ^ 1], xh, (ct + 1) * 16, d, kch, tid);

        const int colbase = ct * 16;
        const int tc = tgt[colbase + q];

        v8f c = {};
#pragma unroll
        for (int t = 0; t < 8; ++t) {
            const _Float16* bp = &bbuf[cur][q][t * 32 + half * 8];
            v8h b0 = *(const v8h*)bp;
            v8h b1 = *(const v8h*)(bp + 16);
            v16h b = __builtin_shufflevector(b0, b1, 0,1,2,3,4,5,6,7,8,9,10,11,12,13,14,15);
            c = __builtin_amdgcn_wmma_f32_16x16x32_f16(false, afrag[t], false, b,
                                                       (short)0, c, false, false);
        }

#pragma unroll
        for (int r = 0; r < 8; ++r) {
            const float v = c[r];
            const int i = rowbase + half * 8 + r;
            const int j = colbase + q;
            const bool same = (tc == trow[r]);
            const bool posm = same && (j != i) && (v < SIM_ONE);
            const bool nsel = !same && (v + margin - mp[r]  > 0.f);
            const bool psel = posm  && (mxg[r] - v + margin > 0.f);
            const float nt = softplus_fast(ALPHA * (v - BASE));
            const float pt = softplus_fast(-BETA * (v - BASE));
            ns[r] += nsel ? nt : 0.f;  nc[r] += nsel ? 1.f : 0.f;
            ps[r] += psel ? pt : 0.f;  pc[r] += psel ? 1.f : 0.f;
        }

        wait_async();
        __syncthreads();
    }

#pragma unroll
    for (int r = 0; r < 8; ++r) {
        float a = ps[r], b = pc[r], e = ns[r], f = nc[r];
#pragma unroll
        for (int m = 8; m >= 1; m >>= 1) {
            a += __shfl_xor(a, m, 16);
            b += __shfl_xor(b, m, 16);
            e += __shfl_xor(e, m, 16);
            f += __shfl_xor(f, m, 16);
        }
        if (q == 0) {
            const size_t idx = (size_t)chunk * n + rowbase + half * 8 + r;
            ps_p[idx] = a; pc_p[idx] = b; ns_p[idx] = e; nc_p[idx] = f;
        }
    }
}

// ---------------- combine pass-2 partials -> loss, prec (deterministic) --------
__global__ void combine_loss(const float* __restrict__ ps_p, const float* __restrict__ pc_p,
                             const float* __restrict__ ns_p, const float* __restrict__ nc_p,
                             float* __restrict__ out, int n) {
    __shared__ float sl[256], si[256];
    const int t = threadIdx.x;
    float L = 0.f, I = 0.f;
    for (int i = t; i < n; i += 256) {
        float a = 0.f, b = 0.f, e = 0.f, f = 0.f;
        for (int c = 0; c < COLCHUNKS; ++c) {
            const size_t idx = (size_t)c * n + i;
            a += ps_p[idx]; b += pc_p[idx]; e += ns_p[idx]; f += nc_p[idx];
        }
        const bool valid = (b >= 1.f) && (f >= 1.f);
        const float rl = (2.0f / BETA) * (a / fmaxf(b, 1.f)) +
                         (2.0f / ALPHA) * (e / fmaxf(f, 1.f));
        L += valid ? rl : 0.f;
        I += valid ? 0.f : 1.f;
    }
    sl[t] = L; si[t] = I;
    __syncthreads();
    for (int stride = 128; stride >= 1; stride >>= 1) {
        if (t < stride) { sl[t] += sl[t + stride]; si[t] += si[t + stride]; }
        __syncthreads();
    }
    if (t == 0) { out[0] = sl[0] / (float)n; out[1] = si[0] / (float)n; }
}

// ---------------- last-row pos/neg mean sims (full f32) ----------------
__global__ void lastrow_kernel(const float* __restrict__ x, const int* __restrict__ tgt,
                               float* __restrict__ out, int n, int d) {
    __shared__ float sp[256], scp[256], sn[256], scn[256];
    const int t = threadIdx.x;
    const int tl = tgt[n - 1];
    const float* xl = x + (size_t)(n - 1) * d;
    float psum = 0.f, pcnt = 0.f, nsum = 0.f, ncnt = 0.f;
    for (int j = t; j < n; j += blockDim.x) {
        const float* xj = x + (size_t)j * d;
        float s = 0.f;
        for (int k = 0; k < d; ++k) s = fmaf(xl[k], xj[k], s);
        const bool same = (tgt[j] == tl);
        if (same && (j != n - 1) && (s < SIM_ONE)) { psum += s; pcnt += 1.f; }
        if (!same)                                  { nsum += s; ncnt += 1.f; }
    }
    sp[t] = psum; scp[t] = pcnt; sn[t] = nsum; scn[t] = ncnt;
    __syncthreads();
    for (int stride = 128; stride >= 1; stride >>= 1) {
        if (t < stride) {
            sp[t]  += sp[t + stride];
            scp[t] += scp[t + stride];
            sn[t]  += sn[t + stride];
            scn[t] += scn[t + stride];
        }
        __syncthreads();
    }
    if (t == 0) {
        out[2] = sp[0] / fmaxf(scp[0], 1.f);
        out[3] = sn[0] / fmaxf(scn[0], 1.f);
    }
}

extern "C" void kernel_launch(void* const* d_in, const int* in_sizes, int n_in,
                              void* d_out, int out_size, void* d_ws, size_t ws_size,
                              hipStream_t stream) {
    const float* x    = (const float*)d_in[0];
    const int*   tgt  = (const int*)d_in[1];
    const float* marg = (const float*)d_in[2];
    float* out = (float*)d_out;

    const int n = in_sizes[1];
    const int d = in_sizes[0] / n;

    // workspace layout
    char* ws = (char*)d_ws;
    size_t off = 0;
    auto alloc = [&](size_t bytes) { void* p = ws + off; off += (bytes + 255) & ~(size_t)255; return p; };
    _Float16* xh  = (_Float16*)alloc((size_t)n * d * sizeof(_Float16));
    float* mp_p   = (float*)alloc((size_t)COLCHUNKS * n * 4);
    float* mx_p   = (float*)alloc((size_t)COLCHUNKS * n * 4);
    float* minpos = (float*)alloc((size_t)n * 4);
    float* maxneg = (float*)alloc((size_t)n * 4);
    float* ps_p   = (float*)alloc((size_t)COLCHUNKS * n * 4);
    float* pc_p   = (float*)alloc((size_t)COLCHUNKS * n * 4);
    float* ns_p   = (float*)alloc((size_t)COLCHUNKS * n * 4);
    float* nc_p   = (float*)alloc((size_t)COLCHUNKS * n * 4);

    const size_t nelem = (size_t)n * d;
    const int rowWGs = n / 128;                 // 8 waves x 16 rows per WG
    const dim3 grid(rowWGs, COLCHUNKS);

    cvt_kernel<<<(unsigned)((nelem + 255) / 256), 256, 0, stream>>>(x, xh, nelem);
    pass1_kernel<<<grid, 256, 0, stream>>>(xh, tgt, mp_p, mx_p, n, d);
    combine_minmax<<<(n + 255) / 256, 256, 0, stream>>>(mp_p, mx_p, minpos, maxneg, n);
    pass2_kernel<<<grid, 256, 0, stream>>>(xh, tgt, minpos, maxneg, marg,
                                           ps_p, pc_p, ns_p, nc_p, n, d);
    combine_loss<<<1, 256, 0, stream>>>(ps_p, pc_p, ns_p, nc_p, out, n);
    lastrow_kernel<<<1, 256, 0, stream>>>(x, tgt, out, n, d);
}
